// RNN_84576495993533
// MI455X (gfx1250) — matrix operand order repaired and checked by reading, not compiled, not run on hardware
//
#include <hip/hip_runtime.h>
#include <hip/hip_bf16.h>
#include <cstddef>

// Problem constants (match reference)
#define Bc 128
#define Tc 1024
#define Ic 128
#define Hc 256
#define Oc 64
#define G4H 1024   // 4*H

typedef __attribute__((ext_vector_type(16))) __bf16 v16bf;
typedef __attribute__((ext_vector_type(8)))  __bf16 v8bf;
typedef __attribute__((ext_vector_type(8)))  float  v8f;

// ---------------------------------------------------------------------------
// Pack kernel: fp32 weight matrix W[N x K] (gates = act @ W^T) -> bf16
// B-matrix fragments laid out exactly as the WMMA lanes consume them.
// Fragment (nt, ks): lane L holds 16 consecutive bf16 (32 bytes):
//   n = nt*16 + (L & 15),  k = ks*32 + (L>>4)*16 + e   (e = element 0..15)
// Linear index: ((nt*Ksteps + ks)*32 + lane)*16 + e
// ---------------------------------------------------------------------------
__global__ void pack_weights_kernel(const float* __restrict__ src,
                                    unsigned short* __restrict__ dst_raw,
                                    int Ntiles, int Ksteps, int K) {
    __bf16* dst = reinterpret_cast<__bf16*>(dst_raw);
    int idx = blockIdx.x * 256 + threadIdx.x;
    int total = Ntiles * Ksteps * 32 * 16;
    if (idx >= total) return;
    int e    = idx & 15;
    int lane = (idx >> 4) & 31;
    int frag = idx >> 9;
    int ks   = frag % Ksteps;
    int nt   = frag / Ksteps;
    int n = nt * 16 + (lane & 15);
    int k = ks * 32 + ((lane >> 4) << 4) + e;
    dst[idx] = (__bf16)src[(size_t)n * K + k];
}

// ---------------------------------------------------------------------------
// Fragment loaders
// ---------------------------------------------------------------------------
// A-matrix fragment (16xK row-major bf16 in LDS, row stride ldk elements):
// lane L: m = L&15; elems 0..15 = k0..k0+15,
// with k0 = ks*32 + (L>=16 ? 16 : 0).   (matches packed B k order)
__device__ inline v16bf load_a_frag(const __bf16* base, int ldk, int lane, int ks) {
    int m  = lane & 15;
    int k0 = ks * 32 + ((lane >> 4) << 4);
    const __bf16* row = base + (size_t)m * ldk + k0;
    union { v16bf v; v8bf h[2]; } u;
    u.h[0] = *reinterpret_cast<const v8bf*>(row);
    u.h[1] = *reinterpret_cast<const v8bf*>(row + 8);
    return u.v;
}

// Packed B fragment: one contiguous 32-byte load per lane (L2-resident).
// `zoff` is an opaque 0 (see loop) so these stay inside the time loop but the
// base pointer keeps its global address space -> global_load_b128 GVS mode.
__device__ inline v16bf load_b_frag(const __bf16* pack, int zoff,
                                    int Ksteps, int nt, int ks, int lane) {
    const __bf16* p = pack + zoff + ((size_t)(nt * Ksteps + ks) * 32 + lane) * 16;
    return *reinterpret_cast<const v16bf*>(p);
}

__device__ inline float sigmoidf(float v) { return 1.0f / (1.0f + __expf(-v)); }

// ---------------------------------------------------------------------------
// Persistent LSTM kernel: one workgroup per 16 batch rows, loops over T.
// 512 threads = 16 waves. Wave w owns gate N-tiles [w*4, w*4+4).
// ---------------------------------------------------------------------------
__global__ __launch_bounds__(512) void lstm_persistent_kernel(
    const float* __restrict__ x,        // [B, T, I] fp32
    const float* __restrict__ b_ih,     // [4H]
    const float* __restrict__ b_hh,     // [4H]
    const float* __restrict__ b_out,    // [O]
    const unsigned short* __restrict__ wih_raw,   // packed bf16, 64x4 frags
    const unsigned short* __restrict__ whh_raw,   // packed bf16, 64x8 frags
    const unsigned short* __restrict__ wout_raw,  // packed bf16,  4x8 frags
    float* __restrict__ out)            // [B, T, O] fp32
{
    const __bf16* wih_p  = reinterpret_cast<const __bf16*>(wih_raw);
    const __bf16* whh_p  = reinterpret_cast<const __bf16*>(whh_raw);
    const __bf16* wout_p = reinterpret_cast<const __bf16*>(wout_raw);

    __shared__ __bf16 h_bf[16][Hc];      //  8 KB  hidden state (A operand)
    __shared__ __bf16 x_bf[16][Ic];      //  4 KB  staged x_t tile (A operand)
    __shared__ float  c_st[16][Hc];      // 16 KB  cell state
    __shared__ float  gates[16][G4H];    // 64 KB  activated gates i|f|g|o

    const int tid   = threadIdx.x;
    const int lane  = tid & 31;
    const int wave  = tid >> 5;
    const int nlane = lane & 15;
    const int bm0   = blockIdx.x * 16;   // first batch row of this WG

    // zero-init h, c
    for (int i = tid; i < 16 * Hc; i += 512) {
        (&h_bf[0][0])[i] = (__bf16)0.0f;
        (&c_st[0][0])[i] = 0.0f;
    }

    // Per-wave loop-invariant bias values (one per owned N-tile).
    float bias0 = b_ih[(wave * 4 + 0) * 16 + nlane] + b_hh[(wave * 4 + 0) * 16 + nlane];
    float bias1 = b_ih[(wave * 4 + 1) * 16 + nlane] + b_hh[(wave * 4 + 1) * 16 + nlane];
    float bias2 = b_ih[(wave * 4 + 2) * 16 + nlane] + b_hh[(wave * 4 + 2) * 16 + nlane];
    float bias3 = b_ih[(wave * 4 + 3) * 16 + nlane] + b_hh[(wave * 4 + 3) * 16 + nlane];
    float boutv = 0.0f;
    if (wave < 4) boutv = b_out[wave * 16 + nlane];

    __syncthreads();

    for (int t = 0; t < Tc; ++t) {
        // Opaque zero, redefined every iteration: keeps the weight-fragment
        // loads inside the time loop (no LICM hoist -> no scratch spills)
        // while preserving the global address space of the base pointers.
        int zoff = 0;
        asm volatile("" : "+v"(zoff));

        // ---- stage x_t tile [16 x I] fp32 -> bf16 LDS (coalesced) ----
        for (int i = tid; i < 16 * Ic; i += 512) {
            int m = i >> 7, k = i & (Ic - 1);
            const float* xp = x + (size_t)(bm0 + m) * Tc * Ic + (size_t)t * Ic + k;
            x_bf[m][k] = (__bf16)(*xp);
            if (t + 1 < Tc) __builtin_prefetch(xp + Ic, 0, 0);  // next step's tile
        }
        __syncthreads();

        // ---- gates = b + x_t @ W_ih^T + h @ W_hh^T  (bf16 WMMA, f32 acc) ----
#pragma unroll 1
        for (int j = 0; j < 4; ++j) {
            const int nt = wave * 4 + j;
            float bj = bias0;
            if (j == 1) bj = bias1;
            else if (j == 2) bj = bias2;
            else if (j == 3) bj = bias3;

            v8f acc;
#pragma unroll
            for (int r = 0; r < 8; ++r) acc[r] = bj;

#pragma unroll
            for (int ks = 0; ks < Ic / 32; ++ks) {            // K = I = 128
                v16bf a = load_a_frag(&x_bf[0][0], Ic, lane, ks);
                v16bf b = load_b_frag(wih_p, zoff, Ic / 32, nt, ks, lane);
                acc = __builtin_amdgcn_wmma_f32_16x16x32_bf16(
                        false, a, false, b, (short)0, acc, false, false);
            }
#pragma unroll
            for (int ks = 0; ks < Hc / 32; ++ks) {            // K = H = 256
                v16bf a = load_a_frag(&h_bf[0][0], Hc, lane, ks);
                v16bf b = load_b_frag(whh_p, zoff, Hc / 32, nt, ks, lane);
                acc = __builtin_amdgcn_wmma_f32_16x16x32_bf16(
                        false, a, false, b, (short)0, acc, false, false);
            }

            // activation: n-tiles [0,16)=i  [16,32)=f  [32,48)=g  [48,64)=o
            const bool is_g = (nt >= 32) && (nt < 48);
#pragma unroll
            for (int r = 0; r < 8; ++r) {
                float v = acc[r];
                v = is_g ? tanhf(v) : sigmoidf(v);
                int m = r + ((lane >> 4) << 3);               // C-layout row
                gates[m][nt * 16 + nlane] = v;
            }
        }
        __syncthreads();

        // ---- elementwise: c = f*c + i*g ; h = o*tanh(c) ----
        for (int i = tid; i < 16 * Hc; i += 512) {
            int m = i >> 8, k = i & (Hc - 1);
            float iA = gates[m][k];
            float fA = gates[m][Hc + k];
            float gA = gates[m][2 * Hc + k];
            float oA = gates[m][3 * Hc + k];
            float cn = fA * c_st[m][k] + iA * gA;
            c_st[m][k] = cn;
            h_bf[m][k] = (__bf16)(oA * tanhf(cn));
        }
        __syncthreads();

        // ---- readout: y = sigmoid(h @ W_out^T + b_out), waves 0..3 ----
        if (wave < 4) {
            const int nt = wave;                               // O = 64 -> 4 tiles
            v8f acc;
#pragma unroll
            for (int r = 0; r < 8; ++r) acc[r] = boutv;
#pragma unroll
            for (int ks = 0; ks < Hc / 32; ++ks) {
                v16bf a = load_a_frag(&h_bf[0][0], Hc, lane, ks);
                v16bf b = load_b_frag(wout_p, zoff, Hc / 32, nt, ks, lane);
                acc = __builtin_amdgcn_wmma_f32_16x16x32_bf16(
                        false, a, false, b, (short)0, acc, false, false);
            }
#pragma unroll
            for (int r = 0; r < 8; ++r) {
                int m = r + ((lane >> 4) << 3);
                out[(size_t)(bm0 + m) * Tc * Oc + (size_t)t * Oc + nt * 16 + nlane]
                    = sigmoidf(acc[r]);
            }
        }
        // No trailing barrier needed: next-iter x_bf writes are fenced by the
        // post-elementwise barrier; h_bf writes are two barriers away.
    }
}

// ---------------------------------------------------------------------------
// Launch
// ---------------------------------------------------------------------------
extern "C" void kernel_launch(void* const* d_in, const int* in_sizes, int n_in,
                              void* d_out, int out_size, void* d_ws, size_t ws_size,
                              hipStream_t stream) {
    const float* x     = (const float*)d_in[0];   // [B,T,I]
    const float* w_ih  = (const float*)d_in[1];   // [4H,I]
    const float* w_hh  = (const float*)d_in[2];   // [4H,H]
    const float* b_ih  = (const float*)d_in[3];   // [4H]
    const float* b_hh  = (const float*)d_in[4];   // [4H]
    const float* w_out = (const float*)d_in[5];   // [O,H]
    const float* b_out = (const float*)d_in[6];   // [O]
    float* out = (float*)d_out;

    // Workspace layout (bf16 elements): packed W_ih | W_hh | W_out
    const int wih_elems  = (G4H / 16) * (Ic / 32) * 32 * 16;  // 131072
    const int whh_elems  = (G4H / 16) * (Hc / 32) * 32 * 16;  // 262144
    const int wout_elems = (Oc  / 16) * (Hc / 32) * 32 * 16;  //  16384
    unsigned short* ws = (unsigned short*)d_ws;
    unsigned short* wih_p  = ws;
    unsigned short* whh_p  = ws + wih_elems;
    unsigned short* wout_p = ws + wih_elems + whh_elems;

    pack_weights_kernel<<<(wih_elems  + 255) / 256, 256, 0, stream>>>(
        w_ih,  wih_p,  G4H / 16, Ic / 32, Ic);
    pack_weights_kernel<<<(whh_elems  + 255) / 256, 256, 0, stream>>>(
        w_hh,  whh_p,  G4H / 16, Hc / 32, Hc);
    pack_weights_kernel<<<(wout_elems + 255) / 256, 256, 0, stream>>>(
        w_out, wout_p, Oc / 16,  Hc / 32, Hc);

    // 8 persistent workgroups (B/16), 512 threads = 16 waves each.
    lstm_persistent_kernel<<<Bc / 16, 512, 0, stream>>>(
        x, b_ih, b_hh, b_out, wih_p, whh_p, wout_p, out);
}
